// MoEExpert_62397284876805
// MI455X (gfx1250) — matrix operand order repaired and checked by reading, not compiled, not run on hardware
//
#include <hip/hip_runtime.h>
#include <stdint.h>

// ---------------------------------------------------------------------------
// SwiGLU FFN for MI455X (gfx1250, wave32, WMMA):
//   inter = silu(x @ w1^T) * (x @ w3^T)      [T, I]  (bf16 in workspace)
//   out   = inter @ w2^T                     [T, H]  (fp32)
// bf16-in / fp32-accumulate WMMA, double-buffered LDS fed by
// global_load_async_to_lds_b128 (ASYNCcnt) issued via inline asm.
// ---------------------------------------------------------------------------

#define TOKENS 4096
#define HIDDEN 4096
#define INTER  14336

// set to 0 to fall back to synchronous VGPR-staged global->LDS copies
#define USE_ASYNC_LDS 1

typedef __attribute__((ext_vector_type(16))) __bf16 v16bf;
typedef __attribute__((ext_vector_type(8)))  __bf16 v8bf;
typedef __attribute__((ext_vector_type(4)))  __bf16 v4bf;
typedef __attribute__((ext_vector_type(8)))  float  v8f;

__device__ __forceinline__ __bf16 f2bf(float f) {
    unsigned int u = __builtin_bit_cast(unsigned int, f);
    unsigned int r = u + 0x7FFFu + ((u >> 16) & 1u);
    unsigned short s = (unsigned short)(r >> 16);
    return __builtin_bit_cast(__bf16, s);
}

// 16-byte global -> LDS copy via the CDNA5 async-DMA path.
// vdst VGPR carries the LDS byte offset (= low 32 bits of the flat shared
// pointer, per ISA 10.2 aperture mapping); vaddr is the 64-bit global addr.
__device__ __forceinline__ void copy16_g2l(__bf16* lds, const __bf16* g) {
#if USE_ASYNC_LDS
    uint32_t loff  = (uint32_t)(uintptr_t)lds;
    uint64_t gaddr = (uint64_t)(uintptr_t)g;
    asm volatile("global_load_async_to_lds_b128 %0, %1, off"
                 :: "v"(loff), "v"(gaddr)
                 : "memory");
#else
    *reinterpret_cast<v8bf*>(lds) = *reinterpret_cast<const v8bf*>(g);
#endif
}

__device__ __forceinline__ void wait_async_lds() {
#if USE_ASYNC_LDS
#if __has_builtin(__builtin_amdgcn_s_wait_asynccnt)
    __builtin_amdgcn_s_wait_asynccnt(0);
#else
    asm volatile("s_wait_asynccnt 0x0" ::: "memory");
#endif
#endif
}

// A fragment (16x32 bf16, ISA layout): base = LDS row pointer (32 elems/row)
// lanes 0-15 (half=0): e0..7=K0..7,  e8..15=K16..23
// lanes16-31 (half=1): e0..7=K8..15, e8..15=K24..31
__device__ __forceinline__ v16bf load_frag_A(const __bf16* base, int half) {
    union { v16bf v; v8bf h[2]; } u;
    u.h[0] = *reinterpret_cast<const v8bf*>(base + 8 * half);
    u.h[1] = *reinterpret_cast<const v8bf*>(base + 16 + 8 * half);
    return u.v;
}

// B fragment (32x16 bf16): lane holds column N, e0..7=K 8*half..+7, e8..15=K 16+8*half..+7
__device__ __forceinline__ v16bf load_frag_B(const __bf16* base, int half) {
    union { v16bf v; v8bf h[2]; } u;
    u.h[0] = *reinterpret_cast<const v8bf*>(base + 8 * half);
    u.h[1] = *reinterpret_cast<const v8bf*>(base + 16 + 8 * half);
    return u.v;
}

__device__ __forceinline__ v8f wmma_bf16(v16bf a, v16bf b, v8f c) {
    return __builtin_amdgcn_wmma_f32_16x16x32_bf16(false, a, false, b,
                                                   (short)0, c, false, false);
}

// ---------------------------------------------------------------------------
// fp32 -> bf16 conversion pre-pass
// ---------------------------------------------------------------------------
__global__ __launch_bounds__(256) void cvt_f32_bf16(const float* __restrict__ src,
                                                    __bf16* __restrict__ dst,
                                                    int n4) {
    int i = blockIdx.x * blockDim.x + threadIdx.x;
    if (i < n4) {
        float4 v = reinterpret_cast<const float4*>(src)[i];
        v4bf o;
        o[0] = f2bf(v.x);
        o[1] = f2bf(v.y);
        o[2] = f2bf(v.z);
        o[3] = f2bf(v.w);
        reinterpret_cast<v4bf*>(dst)[i] = o;
    }
}

// ---------------------------------------------------------------------------
// Tiled GEMM: C[M,N] = A[M,K] * B[N,K]^T, bf16 in, fp32 accumulate.
// 256 threads = 8 waves arranged 4(M) x 2(N); per-wave 32x64 sub-tile.
// 128x128 block tile, K-step 32, double-buffered LDS (async staged).
// FUSED: two B matrices (w1,w3) share the A fragments; silu*mul epilogue.
// ---------------------------------------------------------------------------
template <bool FUSED>
__global__ __launch_bounds__(256) void gemm_swiglu(
    const __bf16* __restrict__ A,    // [Mtotal x K] row-major
    const __bf16* __restrict__ B1,   // [Ntotal x K] row-major
    const __bf16* __restrict__ B2,   // [Ntotal x K] or nullptr
    void* __restrict__ Cout,         // bf16 [M x N] if FUSED else float
    int K, int Ntotal) {

    __shared__ __align__(16) __bf16 As [2][128 * 32];
    __shared__ __align__(16) __bf16 Bs1[2][128 * 32];
    __shared__ __align__(16) __bf16 Bs2[FUSED ? 2 : 1][FUSED ? 128 * 32 : 8];

    const int tid  = threadIdx.x;
    const int lane = tid & 31;
    const int wave = tid >> 5;
    const int half = lane >> 4;
    const int l16  = lane & 15;
    const int wm   = wave & 3;   // 0..3 -> M strip of 32 rows
    const int wn   = wave >> 2;  // 0..1 -> N strip of 64 cols

    const int blockM = blockIdx.y * 128;
    const int blockN = blockIdx.x * 128;

    // staging: 512 chunks of 8 bf16 per tile; this thread moves chunks
    // (r0, kc0) and (r0+64, kc0)
    const int r0  = tid >> 2;
    const int kc0 = (tid & 3) << 3;

    auto stage = [&](int sb, int k0) {
        copy16_g2l(&As[sb][r0 * 32 + kc0],
                   &A[(size_t)(blockM + r0) * K + k0 + kc0]);
        copy16_g2l(&As[sb][(r0 + 64) * 32 + kc0],
                   &A[(size_t)(blockM + r0 + 64) * K + k0 + kc0]);
        copy16_g2l(&Bs1[sb][r0 * 32 + kc0],
                   &B1[(size_t)(blockN + r0) * K + k0 + kc0]);
        copy16_g2l(&Bs1[sb][(r0 + 64) * 32 + kc0],
                   &B1[(size_t)(blockN + r0 + 64) * K + k0 + kc0]);
        if constexpr (FUSED) {
            copy16_g2l(&Bs2[sb][r0 * 32 + kc0],
                       &B2[(size_t)(blockN + r0) * K + k0 + kc0]);
            copy16_g2l(&Bs2[sb][(r0 + 64) * 32 + kc0],
                       &B2[(size_t)(blockN + r0 + 64) * K + k0 + kc0]);
        }
    };

    v8f accG[2][4] = {};
    v8f accU[2][4] = {};

    // prologue: stage first K-slab
    stage(0, 0);
    wait_async_lds();
    __syncthreads();

    int buf = 0;
    for (int k0 = 0; k0 < K; k0 += 32) {
        // kick off DMA of the next K-slab before computing on this one
        if (k0 + 32 < K) stage(buf ^ 1, k0 + 32);

        const __bf16* Ab  = &As[buf][0];
        const __bf16* B1b = &Bs1[buf][0];

        const v16bf a0 = load_frag_A(Ab + (wm * 32 + l16) * 32, half);
        const v16bf a1 = load_frag_A(Ab + (wm * 32 + 16 + l16) * 32, half);

        v16bf b1 = load_frag_B(B1b + (wn * 64 + l16) * 32, half);
        v16bf b2{};
        if constexpr (FUSED)
            b2 = load_frag_B(&Bs2[buf][0] + (wn * 64 + l16) * 32, half);

#pragma unroll
        for (int n = 0; n < 4; ++n) {
            // rotate: issue loads for n+1 before the WMMAs of n
            v16bf nb1 = b1, nb2 = b2;
            if (n < 3) {
                nb1 = load_frag_B(B1b + (wn * 64 + (n + 1) * 16 + l16) * 32, half);
                if constexpr (FUSED)
                    nb2 = load_frag_B(&Bs2[buf][0] + (wn * 64 + (n + 1) * 16 + l16) * 32, half);
            }
            accG[0][n] = wmma_bf16(a0, b1, accG[0][n]);
            accG[1][n] = wmma_bf16(a1, b1, accG[1][n]);
            if constexpr (FUSED) {
                accU[0][n] = wmma_bf16(a0, b2, accU[0][n]);
                accU[1][n] = wmma_bf16(a1, b2, accU[1][n]);
            }
            b1 = nb1;
            b2 = nb2;
        }

        wait_async_lds();   // our async copies into buf^1 are complete
        __syncthreads();    // everyone's copies complete / reads of buf done
        buf ^= 1;
    }

    // ---- epilogue ---------------------------------------------------------
    // C tile layout: VGPR r, lanes0-15 -> M=r, lanes16-31 -> M=r+8; N=l16
    const int rowBase = blockM + wm * 32 + 8 * half;
    const int colBase = blockN + wn * 64 + l16;
    if constexpr (FUSED) {
        __bf16* Cb = reinterpret_cast<__bf16*>(Cout);
#pragma unroll
        for (int s = 0; s < 2; ++s)
#pragma unroll
            for (int n = 0; n < 4; ++n) {
                const int col = colBase + n * 16;
#pragma unroll
                for (int r = 0; r < 8; ++r) {
                    const float g = accG[s][n][r];
                    const float u = accU[s][n][r];
                    const float sv = g / (1.0f + __expf(-g));   // silu(gate)
                    Cb[(size_t)(rowBase + s * 16 + r) * Ntotal + col] = f2bf(sv * u);
                }
            }
    } else {
        float* Cf = reinterpret_cast<float*>(Cout);
#pragma unroll
        for (int s = 0; s < 2; ++s)
#pragma unroll
            for (int n = 0; n < 4; ++n) {
                const int col = colBase + n * 16;
#pragma unroll
                for (int r = 0; r < 8; ++r) {
                    Cf[(size_t)(rowBase + s * 16 + r) * Ntotal + col] = accG[s][n][r];
                }
            }
    }
}

// ---------------------------------------------------------------------------
// Host-side launch
// ---------------------------------------------------------------------------
extern "C" void kernel_launch(void* const* d_in, const int* in_sizes, int n_in,
                              void* d_out, int out_size, void* d_ws, size_t ws_size,
                              hipStream_t stream) {
    (void)in_sizes; (void)n_in; (void)out_size; (void)ws_size;

    const float* x  = (const float*)d_in[0];   // [T, H]
    const float* w1 = (const float*)d_in[1];   // [I, H] gate
    const float* w2 = (const float*)d_in[2];   // [H, I] down
    const float* w3 = (const float*)d_in[3];   // [I, H] up
    float* out = (float*)d_out;                // [T, H]

    // workspace layout (bf16 copies + intermediate), ~480 MB total
    char* p = (char*)d_ws;
    __bf16* xb    = (__bf16*)p; p += (size_t)TOKENS * HIDDEN * 2;
    __bf16* w1b   = (__bf16*)p; p += (size_t)INTER  * HIDDEN * 2;
    __bf16* w3b   = (__bf16*)p; p += (size_t)INTER  * HIDDEN * 2;
    __bf16* w2b   = (__bf16*)p; p += (size_t)HIDDEN * INTER  * 2;
    __bf16* inter = (__bf16*)p; p += (size_t)TOKENS * INTER  * 2;

    // 1) fp32 -> bf16 conversions (bandwidth-bound)
    {
        int n4;
        n4 = (TOKENS * HIDDEN) / 4;
        cvt_f32_bf16<<<(n4 + 255) / 256, 256, 0, stream>>>(x, xb, n4);
        n4 = (INTER * HIDDEN) / 4;
        cvt_f32_bf16<<<(n4 + 255) / 256, 256, 0, stream>>>(w1, w1b, n4);
        cvt_f32_bf16<<<(n4 + 255) / 256, 256, 0, stream>>>(w3, w3b, n4);
        n4 = (HIDDEN * INTER) / 4;
        cvt_f32_bf16<<<(n4 + 255) / 256, 256, 0, stream>>>(w2, w2b, n4);
    }

    // 2) fused gate/up GEMM + SwiGLU epilogue: inter[T,I] (bf16)
    {
        dim3 grid(INTER / 128, TOKENS / 128);   // (112, 32)
        gemm_swiglu<true><<<grid, 256, 0, stream>>>(xb, w1b, w3b, inter,
                                                    HIDDEN, INTER);
    }

    // 3) down projection: out[T,H] = inter @ w2^T (fp32 out)
    {
        dim3 grid(HIDDEN / 128, TOKENS / 128);  // (32, 32)
        gemm_swiglu<false><<<grid, 256, 0, stream>>>(inter, w2b, nullptr, out,
                                                     INTER, HIDDEN);
    }
}
